// MultiHead_5789615915553
// MI455X (gfx1250) — compile-verified, matrix operand order checked by reading, not run on hardware
//
#include <hip/hip_runtime.h>
#include <hip/hip_bf16.h>

// ---------- types ----------
typedef __attribute__((ext_vector_type(16))) __bf16 v16bf;
typedef __attribute__((ext_vector_type(8)))  float  v8f;
typedef __attribute__((ext_vector_type(4)))  unsigned int u32x4;
typedef __attribute__((ext_vector_type(8)))  int i32x8;
typedef __attribute__((ext_vector_type(4)))  int i32x4;

union Frag {
  v16bf bf;
  u32x4 u4[2];
  unsigned int u[8];
};

__device__ __forceinline__ unsigned short f2bf(float f) {
  unsigned int u = __float_as_uint(f);
  u += 0x7FFFu + ((u >> 16) & 1u);        // round-to-nearest-even
  return (unsigned short)(u >> 16);
}
__device__ __forceinline__ unsigned int pack2bf(float lo, float hi) {
  return (unsigned int)f2bf(lo) | ((unsigned int)f2bf(hi) << 16);
}

// generic -> LDS(AS3) byte address for TDM descriptors
typedef const __attribute__((address_space(3))) void* lds_cptr_t;
__device__ __forceinline__ unsigned ldsByteAddr(const void* p) {
  return (unsigned)(unsigned long long)(lds_cptr_t)p;
}

// ---------- TDM: 2D tile (bf16) global -> LDS with padded rows ----------
// cfg packs D# group1 word0: data_size=2B, pad_enable, pad_interval, pad_amount
#define TDM_CFG_ROW32 ((1u << 16) | (1u << 20) | (3u << 22) | (3u << 25)) // 16-dw rows + 4-dw pad
#define TDM_CFG_ROW64 ((1u << 16) | (1u << 20) | (4u << 22) | (3u << 25)) // 32-dw rows + 4-dw pad

__device__ __forceinline__ void tdm_load_2d(
    unsigned lds, const void* gaddr,
    unsigned tensor_d0, unsigned tensor_d1,
    unsigned tile_d0, unsigned tile_d1,
    unsigned stride_elems, unsigned cfg)
{
  unsigned long long ga = (unsigned long long)gaddr;
  u32x4 g0;
  g0[0] = 1u;                                             // count=1, user descriptor
  g0[1] = lds;                                            // lds_addr
  g0[2] = (unsigned)(ga & 0xFFFFFFFFull);                 // global_addr[31:0]
  g0[3] = (unsigned)((ga >> 32) & 0x1FFFFFFull) | (2u << 30); // addr[56:32] | type=2
  i32x8 g1;
  g1[0] = (int)cfg;
  g1[1] = (int)((tensor_d0 & 0xFFFFu) << 16);             // abar=0 | tensor_dim0 lo16
  g1[2] = (int)((tensor_d0 >> 16) | ((tensor_d1 & 0xFFFFu) << 16));
  g1[3] = (int)((tensor_d1 >> 16) | (tile_d0 << 16));     // tensor_dim1 hi | tile_dim0
  g1[4] = (int)(tile_d1 & 0xFFFFu);                       // tile_dim1 | tile_dim2=0
  g1[5] = (int)stride_elems;                              // tensor_dim0_stride lo32
  g1[6] = 0;                                              // stride hi16 | dim1_stride lo16
  g1[7] = 0;
  i32x4 z4 = {0, 0, 0, 0};
  i32x8 z8 = {0, 0, 0, 0, 0, 0, 0, 0};
  __builtin_amdgcn_tensor_load_to_lds(g0, g1, z4, z4, z8, 0);
}

// ---------- f32 -> bf16 conversion ----------
__global__ void cvt_f32_bf16(const float* __restrict__ src,
                             unsigned short* __restrict__ dst, int n2) {
  int i = blockIdx.x * blockDim.x + threadIdx.x;
  if (i < n2) {
    float a = src[2 * i], b = src[2 * i + 1];
    reinterpret_cast<unsigned int*>(dst)[i] = pack2bf(a, b);
  }
}

// ---------- WMMA GEMM:  C[M,N] = A[M,K] * W[N,K]^T  (bf16 in, f32 out) ----------
#define LDT 40   // padded LDS row stride (ushorts): 32 data + 8 pad (matches TDM pad cfg)
__global__ __launch_bounds__(128) void gemm_bf16(
    const unsigned short* __restrict__ A,
    const unsigned short* __restrict__ W,
    float* __restrict__ C,
    int M, int N, int K)
{
  __shared__ __align__(16) unsigned short As[2][128 * LDT];
  __shared__ __align__(16) unsigned short Bs[2][128 * LDT];
  const int tid  = threadIdx.x;
  const int lane = tid & 31;
  const int wave = tid >> 5;
  const int wm = (wave >> 1) * 64;
  const int wn = (wave & 1) * 64;
  const int bm = blockIdx.y * 128;
  const int bn = blockIdx.x * 128;
  const int r15  = lane & 15;
  const int ch   = (lane >> 4) * 8;
  const int rowh = (lane >> 4) * 8;

  v8f acc[4][4];
  v8f zero = {};
  for (int i = 0; i < 4; ++i)
    for (int j = 0; j < 4; ++j) acc[i][j] = zero;

  // prologue: TDM-stage first k-tile into buffer 0 (wave 0 drives the DMA)
  if (wave == 0) {
    tdm_load_2d(ldsByteAddr(&As[0][0]), A + (size_t)bm * K, K, M, 32, 128, K, TDM_CFG_ROW32);
    tdm_load_2d(ldsByteAddr(&Bs[0][0]), W + (size_t)bn * K, K, N, 32, 128, K, TDM_CFG_ROW32);
  }

  int bufi = 0;
  for (int kt = 0; kt < K; kt += 32, bufi ^= 1) {
    if (wave == 0) {
      if (kt + 32 < K) {   // issue next tile into other buffer, wait for current pair
        tdm_load_2d(ldsByteAddr(&As[bufi ^ 1][0]), A + (size_t)bm * K + kt + 32,
                    K, M, 32, 128, K, TDM_CFG_ROW32);
        tdm_load_2d(ldsByteAddr(&Bs[bufi ^ 1][0]), W + (size_t)bn * K + kt + 32,
                    K, N, 32, 128, K, TDM_CFG_ROW32);
        __builtin_amdgcn_s_wait_tensorcnt(2);
      } else {
        __builtin_amdgcn_s_wait_tensorcnt(0);
      }
    }
    __syncthreads();

    Frag a[4], b[4];
    for (int i = 0; i < 4; ++i) {
      const unsigned short* p = &As[bufi][0] + (wm + i * 16 + r15) * LDT + ch;
      a[i].u4[0] = *reinterpret_cast<const u32x4*>(p);
      a[i].u4[1] = *reinterpret_cast<const u32x4*>(p + 16);
    }
    for (int j = 0; j < 4; ++j) {
      const unsigned short* p = &Bs[bufi][0] + (wn + j * 16 + r15) * LDT + ch;
      b[j].u4[0] = *reinterpret_cast<const u32x4*>(p);
      b[j].u4[1] = *reinterpret_cast<const u32x4*>(p + 16);
    }
    for (int i = 0; i < 4; ++i)
      for (int j = 0; j < 4; ++j)
        acc[i][j] = __builtin_amdgcn_wmma_f32_16x16x32_bf16(
            false, a[i].bf, false, b[j].bf, (short)0, acc[i][j], false, false);
    __syncthreads();
  }

  for (int i = 0; i < 4; ++i)
    for (int j = 0; j < 4; ++j) {
      int n = bn + wn + j * 16 + r15;
      for (int v = 0; v < 8; ++v) {
        int m = bm + wm + i * 16 + v + rowh;
        C[(size_t)m * N + n] = acc[i][j][v];
      }
    }
}

// ---------- RoPE on Q,K (f32 -> bf16 [B,H,S,dk]) + V -> V^T bf16 [B,H,dk,S] ----------
__global__ void rope_cvt(const float* __restrict__ Qf, const float* __restrict__ Kf,
                         const float* __restrict__ Vf,
                         unsigned short* __restrict__ Qb, unsigned short* __restrict__ Kb,
                         unsigned short* __restrict__ Vt)
{
  int t  = blockIdx.x * blockDim.x + threadIdx.x;   // 2,097,152 threads
  int d2 = t & 31;   t >>= 5;
  int h  = t & 15;   t >>= 4;
  int s  = t & 2047; t >>= 11;
  int b  = t;
  if (b >= 2) return;
  float inv_freq = __expf(-(float)(2 * d2) * (9.210340371976184f / 64.0f));
  float ang = (float)s * inv_freq;
  float sn, cs; __sincosf(ang, &sn, &cs);
  size_t src  = ((size_t)(b * 2048 + s)) * 1024 + h * 64 + 2 * d2;
  size_t dstq = (((size_t)(b * 16 + h)) * 2048 + s) * 64 + 2 * d2;
  float q1 = Qf[src], q2 = Qf[src + 1];
  reinterpret_cast<unsigned int*>(Qb)[dstq >> 1] = pack2bf(q1 * cs - q2 * sn, q1 * sn + q2 * cs);
  float k1 = Kf[src], k2 = Kf[src + 1];
  reinterpret_cast<unsigned int*>(Kb)[dstq >> 1] = pack2bf(k1 * cs - k2 * sn, k1 * sn + k2 * cs);
  size_t dv = (((size_t)(b * 16 + h)) * 64 + 2 * d2) * 2048 + s;
  Vt[dv]        = f2bf(Vf[src]);
  Vt[dv + 2048] = f2bf(Vf[src + 1]);
}

// ---------- Flash attention (causal, online softmax), WMMA + TDM staging ----------
#define LKT 72   // padded LDS row stride (ushorts): 64 data + 8 pad (matches TDM pad cfg)
__global__ __launch_bounds__(128) void attn_fwd(
    const unsigned short* __restrict__ Qb,   // [B,H,S,64]
    const unsigned short* __restrict__ Kb,   // [B,H,S,64]
    const unsigned short* __restrict__ Vt,   // [B,H,64,S]
    unsigned short* __restrict__ Ob)         // [B,S,H*64]
{
  __shared__ __align__(16) unsigned short Ks[2][64 * LKT];
  __shared__ __align__(16) unsigned short Vs[2][64 * LKT];
  __shared__ __align__(16) unsigned short Ps[4 * 16 * LKT];
  const int tid  = threadIdx.x;
  const int lane = tid & 31;
  const int wave = tid >> 5;
  const int qb = blockIdx.x & 31;
  const int h  = (blockIdx.x >> 5) & 15;
  const int b  = blockIdx.x >> 9;
  const int r15  = lane & 15;
  const int ch   = (lane >> 4) * 8;
  const int rowh = (lane >> 4) * 8;

  const unsigned short* Qh = Qb + ((size_t)(b * 16 + h)) * 2048 * 64;
  const unsigned short* Kh = Kb + ((size_t)(b * 16 + h)) * 2048 * 64;
  const unsigned short* Vh = Vt + ((size_t)(b * 16 + h)) * 64 * 2048;

  // Q fragments: 16 query rows per wave, dk=64 -> 2 A-frags, kept in VGPRs
  Frag qf[2];
  {
    const unsigned short* qp = Qh + (size_t)(qb * 64 + wave * 16 + r15) * 64;
    for (int kc = 0; kc < 2; ++kc) {
      qf[kc].u4[0] = *reinterpret_cast<const u32x4*>(qp + kc * 32 + ch);
      qf[kc].u4[1] = *reinterpret_cast<const u32x4*>(qp + kc * 32 + ch + 16);
    }
  }

  v8f acc[4]; v8f zero = {};
  for (int j = 0; j < 4; ++j) acc[j] = zero;
  float mrow[8], lrow[8];
  for (int v = 0; v < 8; ++v) { mrow[v] = -__builtin_inff(); lrow[v] = 0.f; }

  unsigned short* Pw = Ps + wave * 16 * LKT;

  // prologue: TDM-stage K/V^T tiles for key-block 0
  if (wave == 0) {
    tdm_load_2d(ldsByteAddr(&Ks[0][0]), Kh, 64, 2048, 64, 64, 64,   TDM_CFG_ROW64);
    tdm_load_2d(ldsByteAddr(&Vs[0][0]), Vh, 2048, 64, 64, 64, 2048, TDM_CFG_ROW64);
  }

  int bufi = 0;
  for (int kb = 0; kb <= qb; ++kb, bufi ^= 1) {
    if (wave == 0) {
      if (kb < qb) {       // stage next key block into other buffer
        tdm_load_2d(ldsByteAddr(&Ks[bufi ^ 1][0]), Kh + (size_t)(kb + 1) * 64 * 64,
                    64, 2048, 64, 64, 64, TDM_CFG_ROW64);
        tdm_load_2d(ldsByteAddr(&Vs[bufi ^ 1][0]), Vh + (size_t)(kb + 1) * 64,
                    2048, 64, 64, 64, 2048, TDM_CFG_ROW64);
        __builtin_amdgcn_s_wait_tensorcnt(2);
      } else {
        __builtin_amdgcn_s_wait_tensorcnt(0);
      }
    }
    __syncthreads();

    // scores: 16 (q) x 64 (k) per wave
    v8f sc[4];
    for (int j = 0; j < 4; ++j) {
      v8f s = zero;
      for (int kc = 0; kc < 2; ++kc) {
        Frag kf;
        const unsigned short* kp = &Ks[bufi][0] + (j * 16 + r15) * LKT + kc * 32 + ch;
        kf.u4[0] = *reinterpret_cast<const u32x4*>(kp);
        kf.u4[1] = *reinterpret_cast<const u32x4*>(kp + 16);
        s = __builtin_amdgcn_wmma_f32_16x16x32_bf16(
            false, qf[kc].bf, false, kf.bf, (short)0, s, false, false);
      }
      sc[j] = s;
    }

    // scale + causal mask (diag block) + online softmax update
    const bool diag = (kb == qb);
    float mnew[8], rsum[8];
    for (int v = 0; v < 8; ++v) {
      float t = -__builtin_inff();
      for (int j = 0; j < 4; ++j) {
        float x = sc[j][v] * 0.125f;          // 1/sqrt(64)
        if (diag) {
          int kg = j * 16 + r15;
          int qg = wave * 16 + v + rowh;
          if (kg > qg) x = -3.0e38f;
        }
        sc[j][v] = x;
        t = fmaxf(t, x);
      }
      for (int msk = 1; msk < 16; msk <<= 1) t = fmaxf(t, __shfl_xor(t, msk, 32));
      mnew[v] = fmaxf(mrow[v], t);
      float scale = __expf(mrow[v] - mnew[v]);
      lrow[v] *= scale;
      for (int j = 0; j < 4; ++j) acc[j][v] *= scale;
      mrow[v] = mnew[v];
      rsum[v] = 0.f;
    }
    for (int j = 0; j < 4; ++j)
      for (int v = 0; v < 8; ++v) {
        float p = __expf(sc[j][v] - mnew[v]);
        rsum[v] += p;
        Pw[(v + rowh) * LKT + j * 16 + r15] = f2bf(p);   // C-layout -> LDS for relayout
      }
    for (int v = 0; v < 8; ++v) {
      float rs = rsum[v];
      for (int msk = 1; msk < 16; msk <<= 1) rs += __shfl_xor(rs, msk, 32);
      lrow[v] += rs;
    }

    // O += P (16x64) x V (64x64): P re-read from LDS in A-layout, B from V^T tile
    for (int kc = 0; kc < 2; ++kc) {
      Frag pf;
      const unsigned short* pp = Pw + r15 * LKT + kc * 32 + ch;
      pf.u4[0] = *reinterpret_cast<const u32x4*>(pp);
      pf.u4[1] = *reinterpret_cast<const u32x4*>(pp + 16);
      for (int j = 0; j < 4; ++j) {
        Frag vf;
        const unsigned short* vp = &Vs[bufi][0] + (j * 16 + r15) * LKT + kc * 32 + ch;
        vf.u4[0] = *reinterpret_cast<const u32x4*>(vp);
        vf.u4[1] = *reinterpret_cast<const u32x4*>(vp + 16);
        acc[j] = __builtin_amdgcn_wmma_f32_16x16x32_bf16(
            false, pf.bf, false, vf.bf, (short)0, acc[j], false, false);
      }
    }
    __syncthreads();
  }

  // epilogue: normalize and emit O bf16 in [B,S,H*64]
  for (int v = 0; v < 8; ++v) {
    float inv = 1.0f / lrow[v];
    int srow = b * 2048 + qb * 64 + wave * 16 + v + rowh;
    for (int j = 0; j < 4; ++j)
      Ob[(size_t)srow * 1024 + h * 64 + j * 16 + r15] = f2bf(acc[j][v] * inv);
  }
}

// ---------- host-side orchestration ----------
extern "C" void kernel_launch(void* const* d_in, const int* in_sizes, int n_in,
                              void* d_out, int out_size, void* d_ws, size_t ws_size,
                              hipStream_t stream) {
  (void)in_sizes; (void)n_in; (void)out_size; (void)ws_size;
  const float* x  = (const float*)d_in[0];
  const float* Wq = (const float*)d_in[1];
  const float* Wk = (const float*)d_in[2];
  const float* Wv = (const float*)d_in[3];
  const float* Wo = (const float*)d_in[4];
  float* out = (float*)d_out;

  const int B = 2, S = 2048, D = 1024;
  const int MS = B * S;          // 4096

  char* ws = (char*)d_ws;
  unsigned short* xbf  = (unsigned short*)(ws + 0);          //  8 MB
  unsigned short* wqbf = (unsigned short*)(ws + (8u  << 20));//  2 MB each
  unsigned short* wkbf = (unsigned short*)(ws + (10u << 20));
  unsigned short* wvbf = (unsigned short*)(ws + (12u << 20));
  unsigned short* wobf = (unsigned short*)(ws + (14u << 20));
  float* qf32 = (float*)(ws + (16u << 20));                  // 16 MB each
  float* kf32 = (float*)(ws + (32u << 20));
  float* vf32 = (float*)(ws + (48u << 20));
  unsigned short* qbf  = (unsigned short*)(ws + (64u << 20));//  8 MB each
  unsigned short* kbf  = (unsigned short*)(ws + (72u << 20));
  unsigned short* vtbf = (unsigned short*)(ws + (80u << 20));
  unsigned short* obf  = (unsigned short*)(ws + (88u << 20));

  // 1) f32 -> bf16 conversions
  {
    int n2x = MS * D / 2;        // 2,097,152
    cvt_f32_bf16<<<(n2x + 255) / 256, 256, 0, stream>>>(x, xbf, n2x);
    int n2w = D * D / 2;         // 524,288
    cvt_f32_bf16<<<(n2w + 255) / 256, 256, 0, stream>>>(Wq, wqbf, n2w);
    cvt_f32_bf16<<<(n2w + 255) / 256, 256, 0, stream>>>(Wk, wkbf, n2w);
    cvt_f32_bf16<<<(n2w + 255) / 256, 256, 0, stream>>>(Wv, wvbf, n2w);
    cvt_f32_bf16<<<(n2w + 255) / 256, 256, 0, stream>>>(Wo, wobf, n2w);
  }

  // 2) Q/K/V projections: [4096,1024] = xbf * W^T
  dim3 ggrid(D / 128, MS / 128);   // (8, 32)
  gemm_bf16<<<ggrid, 128, 0, stream>>>(xbf, wqbf, qf32, MS, D, D);
  gemm_bf16<<<ggrid, 128, 0, stream>>>(xbf, wkbf, kf32, MS, D, D);
  gemm_bf16<<<ggrid, 128, 0, stream>>>(xbf, wvbf, vf32, MS, D, D);

  // 3) RoPE + relayout (Q,K -> [B,H,S,dk] bf16; V -> [B,H,dk,S] bf16)
  {
    int nthr = B * S * 16 * 32;    // 2,097,152
    rope_cvt<<<(nthr + 255) / 256, 256, 0, stream>>>(qf32, kf32, vf32, qbf, kbf, vtbf);
  }

  // 4) causal flash attention -> O bf16 [B,S,H*dk]
  attn_fwd<<<B * 16 * (S / 64), 128, 0, stream>>>(qbf, kbf, vtbf, obf);

  // 5) output projection into d_out (fp32)
  gemm_bf16<<<ggrid, 128, 0, stream>>>(obf, wobf, out, MS, D, D);
}
